// MoE_13245679141624
// MI455X (gfx1250) — matrix operand order repaired and checked by reading, not compiled, not run on hardware
//
#include <hip/hip_runtime.h>
#include <hip/hip_bf16.h>

// ---------------------------------------------------------------------------
// MoE (noisy top-2 gating + per-expert FFN) for gfx1250 / MI455X.
//   B=2, S=2048 -> N=4096 tokens, D=768, H=3072, E=8, TOP_K=2.
// Pipeline:
//   k0: zero out[] and per-expert counts
//   k1: gating (wave-per-token), top-2 + softmax, bucket (token,prob) per expert
//   k2: per-expert FFN tiles on v_wmma_f32_16x16x32_bf16 (all 8 waves busy in
//       both GEMMs). x tile staged via TDM gather (tensor_load_to_lds,
//       TENSORcnt) + async-to-LDS loads (ASYNCcnt); atomicAdd scatter.
// ---------------------------------------------------------------------------

typedef __attribute__((ext_vector_type(16))) __bf16 v16bf;
typedef __attribute__((ext_vector_type(8)))  float  v8f;
typedef __attribute__((ext_vector_type(4)))  unsigned int v4u;
typedef __attribute__((ext_vector_type(8)))  unsigned int v8u;

#define NTOK 4096
#define D_   768
#define H_   3072
#define E_   8

#define XS_STRIDE  776   // halves per x row in LDS (768 + pad, keeps 16B align)
#define W2T_STRIDE 40    // halves per transposed-W2 row (32 + pad)
#define W1T_STRIDE 136   // halves per transposed-W1 row (128 + pad)
#define HID_STRIDE 72    // halves per hid row (64 + pad)

// dynamic-LDS layout (units: unsigned short)
#define XS_OFF   0
#define W2T_OFF  (XS_OFF  + 32  * XS_STRIDE)    // 24832  (also fp32 x-stage scratch)
#define W1T_OFF  (W2T_OFF + 768 * W2T_STRIDE)   // 55552
#define HID_OFF  (W1T_OFF + 64  * W1T_STRIDE)   // 64256
#define META_OFF (HID_OFF + 32  * HID_STRIDE)   // 66560
#define SMEM_BYTES (META_OFF * 2 + 32 * 4 + 32 * 4)  // + probs[32] + toks[32]

__device__ __forceinline__ unsigned short f2bf(float f) {
  union { float f; unsigned u; } c; c.f = f;
  unsigned u = c.u;
  return (unsigned short)((u + 0x7FFFu + ((u >> 16) & 1u)) >> 16);  // RNE
}

// ---------------------------------------------------------------------------
__global__ void zero_kernel(float* __restrict__ out, int n, int* __restrict__ counts) {
  int i = blockIdx.x * blockDim.x + threadIdx.x;
  if (i < n) out[i] = 0.0f;
  if (i < E_) counts[i] = 0;
}

// ---------------------------------------------------------------------------
// Gating: one wave32 per token.
__global__ __launch_bounds__(256) void gating_kernel(
    const float* __restrict__ x, const float* __restrict__ noise,
    const float* __restrict__ Wg, const float* __restrict__ bg,
    const float* __restrict__ Wn, const float* __restrict__ bn,
    int* __restrict__ counts, int* __restrict__ btok, float* __restrict__ bprob) {
  const int wave = threadIdx.x >> 5;
  const int lane = threadIdx.x & 31;
  const int token = blockIdx.x * 8 + wave;
  if (token >= NTOK) return;

  float g[E_] = {}, nlog[E_] = {};
  const float* xr = x + (size_t)token * D_;
  for (int d = lane; d < D_; d += 32) {
    const float xv = xr[d];
    const float* wg = Wg + (size_t)d * E_;
    const float* wn = Wn + (size_t)d * E_;
#pragma unroll
    for (int e = 0; e < E_; ++e) { g[e] += xv * wg[e]; nlog[e] += xv * wn[e]; }
  }
#pragma unroll
  for (int e = 0; e < E_; ++e) {
#pragma unroll
    for (int off = 16; off; off >>= 1) {
      g[e]    += __shfl_xor(g[e], off, 32);
      nlog[e] += __shfl_xor(nlog[e], off, 32);
    }
  }
  if (lane == 0) {
    float h[E_];
    const float* nz = noise + (size_t)token * E_;
#pragma unroll
    for (int e = 0; e < E_; ++e) {
      const float z = nlog[e] + bn[e];
      const float sp = (z > 20.0f) ? z : log1pf(expf(z));  // softplus
      h[e] = (g[e] + bg[e]) + nz[e] * sp;
    }
    int i0 = 0;
#pragma unroll
    for (int e = 1; e < E_; ++e) if (h[e] > h[i0]) i0 = e;
    int i1 = (i0 == 0) ? 1 : 0;
#pragma unroll
    for (int e = 0; e < E_; ++e) if (e != i0 && h[e] > h[i1]) i1 = e;
    const float p0 = 1.0f / (1.0f + expf(h[i1] - h[i0]));  // 2-way softmax
    const float p1 = 1.0f - p0;
    int s0 = atomicAdd(&counts[i0], 1);
    btok[i0 * NTOK + s0] = token; bprob[i0 * NTOK + s0] = p0;
    int s1 = atomicAdd(&counts[i1], 1);
    btok[i1 * NTOK + s1] = token; bprob[i1 * NTOK + s1] = p1;
  }
}

// ---------------------------------------------------------------------------
// Expert FFN: grid = (128 tiles, 8 experts), block = 256 (8 waves).
// Tile = 32 tokens x 768 outputs. H chunked by 64: GEMM1 produces hid[32x64]
// (8 WMMA tiles -> one per wave), GEMM2 consumes it in two K=32 sub-steps
// (12 WMMA per wave each). Both GEMMs keep all 8 waves on WMMA.
__global__ __launch_bounds__(256) void moe_ffn_kernel(
    const float* __restrict__ x,
    const float* __restrict__ W1, const float* __restrict__ b1,
    const float* __restrict__ W2, const float* __restrict__ b2,
    const int* __restrict__ counts, const int* __restrict__ btok,
    const float* __restrict__ bprob, float* __restrict__ out) {
  extern __shared__ unsigned short smem[];
  unsigned short* xs  = smem + XS_OFF;    // [32][XS_STRIDE]  bf16 x tile
  unsigned short* w2t = smem + W2T_OFF;   // [768][W2T_STRIDE] bf16 W2 slice (transposed)
  unsigned short* w1t = smem + W1T_OFF;   // [64][W1T_STRIDE]  bf16 W1 slice (transposed)
  unsigned short* hid = smem + HID_OFF;   // [32][HID_STRIDE]  bf16 hidden chunk
  float* probs_s = (float*)(smem + META_OFF);
  int*   toks_s  = (int*)(probs_s + 32);

  const int e   = blockIdx.y;
  const int m0  = blockIdx.x * 32;
  const int cnt = counts[e];
  if (m0 >= cnt) return;                  // uniform early-exit
  const int tid  = threadIdx.x;
  const int wave = tid >> 5, lane = tid & 31;

  if (tid < 32) {
    const int r = m0 + tid;
    toks_s[tid]  = (r < cnt) ? btok[e * NTOK + r]  : 0;
    probs_s[tid] = (r < cnt) ? bprob[e * NTOK + r] : 0.0f;
  }
  __syncthreads();

  // --- stage x rows (raw fp32) into LDS scratch (w2t region, 48KB), then
  // convert to bf16 xs.  Half 0 uses the Tensor Data Mover in gather mode
  // (one DMA pulls 16 routed token rows, TENSORcnt).  Half 1 uses per-lane
  // global_load_async_to_lds_b128 (ASYNCcnt).
  {
    float* xstage = (float*)w2t;
    const unsigned stage_base = (unsigned)(W2T_OFF * 2);  // LDS byte offset

    // ---- half 0: TDM gather --------------------------------------------
    {
      unsigned idx[8];
#pragma unroll
      for (int wq = 0; wq < 8; ++wq) {   // pack 16 x 16-bit row indices
        const unsigned lo = (unsigned)__builtin_amdgcn_readfirstlane(toks_s[2 * wq]);
        const unsigned hi = (unsigned)__builtin_amdgcn_readfirstlane(toks_s[2 * wq + 1]);
        idx[wq] = (lo & 0xFFFFu) | (hi << 16);
      }
      const unsigned long long ga = (unsigned long long)(const void*)x;
      v4u g0;                                        // D# group 0 (ISA 8.3)
      g0[0] = 0x80000001u;                           // count=1, gather, 16b idx
      g0[1] = stage_base;                            // lds_addr (bytes)
      g0[2] = (unsigned)(ga & 0xFFFFFFFFu);          // global_addr[31:0]
      g0[3] = (unsigned)((ga >> 32) & 0x1FFFFFFu)    // global_addr[56:32]
              | 0x80000000u;                         // type=2 ("image")
      v8u g1;                                        // D# group 1 (ISA 8.4)
      g1[0] = 0x00020000u;                           // data_size=4B, no mask
      g1[1] = (unsigned)D_ << 16;                    // tensor_dim0[15:0]=768
      g1[2] = (unsigned)NTOK << 16;                  // tensor_dim1[15:0]=4096
      g1[3] = (unsigned)D_ << 16;                    // tile_dim0 = 768
      g1[4] = 16u;                                   // tile_dim1 = #indices
      g1[5] = (unsigned)D_;                          // tensor_dim0_stride = 768
      g1[6] = 0u;
      g1[7] = 0u;
      v4u g2, g3;                                    // row indices (gather)
      g2[0] = idx[0]; g2[1] = idx[1]; g2[2] = idx[2]; g2[3] = idx[3];
      g3[0] = idx[4]; g3[1] = idx[5]; g3[2] = idx[6]; g3[3] = idx[7];
      if (wave == 0) {                               // one DMA per workgroup
        asm volatile("tensor_load_to_lds %0, %1, %2, %3"
                     :: "s"(g0), "s"(g1), "s"(g2), "s"(g3) : "memory");
        __builtin_amdgcn_s_wait_tensorcnt(0);
      }
      __syncthreads();
      for (int i = tid; i < 16 * D_; i += 256) {
        const int r = i / D_, c = i % D_;
        xs[r * XS_STRIDE + c] = f2bf(xstage[r * D_ + c]);
      }
      __syncthreads();
    }
    // ---- half 1: per-lane async-to-LDS b128 ----------------------------
    {
      for (int i = tid; i < 16 * (D_ / 4); i += 256) {
        const int r = i / (D_ / 4), c4 = i % (D_ / 4);
        const unsigned long long gaddr = (unsigned long long)(const void*)
            (x + (size_t)toks_s[16 + r] * D_ + (size_t)c4 * 4);
        const unsigned ldsoff = stage_base + (unsigned)(r * (D_ * 4) + c4 * 16);
        asm volatile("global_load_async_to_lds_b128 %0, %1, off"
                     :: "v"(ldsoff), "v"(gaddr) : "memory");
      }
      asm volatile("s_wait_asynccnt 0" ::: "memory");
      __syncthreads();
      for (int i = tid; i < 16 * D_; i += 256) {
        const int r = i / D_, c = i % D_;
        xs[(16 + r) * XS_STRIDE + c] = f2bf(xstage[r * D_ + c]);
      }
      __syncthreads();
    }
  }

  const float* W1e = W1 + (size_t)e * D_ * H_;
  const float* W2e = W2 + (size_t)e * H_ * D_;

  v8f acc2[12];
#pragma unroll
  for (int j = 0; j < 12; ++j) acc2[j] = (v8f){0, 0, 0, 0, 0, 0, 0, 0};

  const int mt = wave & 1;   // token-half (M tile) this wave owns in GEMM2
  const int cg = wave >> 1;  // column-group 0..3 (interleaved 16-wide N tiles)
  const int m1 = wave & 1;   // GEMM1 tile of hid[32x64]: (m1, n1), 8 tiles/8 waves
  const int n1 = wave >> 1;

  for (int kh = 0; kh < H_; kh += 64) {
    // ---- GEMM1: hid[32x64] = relu(xs[32x768] @ W1e[:, kh:kh+64] + b1) ----
    v8f acc1 = (v8f){0, 0, 0, 0, 0, 0, 0, 0};
    for (int kko = 0; kko < D_; kko += 128) {
      // stage W1 slice transposed: w1t[c][r] = W1e[kko+r][kh+c], 128K x 64N
      for (int i = tid; i < 128 * 64; i += 256) {
        const int r = i >> 6, c = i & 63;
        w1t[c * W1T_STRIDE + r] = f2bf(W1e[(size_t)(kko + r) * H_ + kh + c]);
      }
      if (kko + 128 < D_)
        __builtin_prefetch(&W1e[(size_t)(kko + 128) * H_ + kh], 0, 1);
      __syncthreads();
#pragma unroll
      for (int kk2 = 0; kk2 < 128; kk2 += 32) {
        union { v16bf v; v4u u[2]; } A, Bf;
        {   // A operand (16x32 bf16), wave32 layout, rows of xs
          const int row = m1 * 16 + (lane & 15);
          const unsigned short* p =
              &xs[row * XS_STRIDE + kko + kk2 + ((lane < 16) ? 0 : 8)];
          A.u[0] = *(const v4u*)(p);
          A.u[1] = *(const v4u*)(p + 16);
        }
        {   // B operand (32x16 bf16), columns from transposed W1 slice
          const int n = n1 * 16 + (lane & 15);
          const unsigned short* p =
              &w1t[n * W1T_STRIDE + kk2 + ((lane < 16) ? 0 : 8)];
          Bf.u[0] = *(const v4u*)(p);
          Bf.u[1] = *(const v4u*)(p + 16);
        }
        acc1 = __builtin_amdgcn_wmma_f32_16x16x32_bf16(
            false, A.v, false, Bf.v, (short)0, acc1, false, false);
      }
      __syncthreads();
    }
    // bias + ReLU + store hid chunk (C/D layout -> row-major bf16 LDS)
    {
      const int col = n1 * 16 + (lane & 15);
      const float bv = b1[(size_t)e * H_ + kh + col];
      const int rbase = m1 * 16 + ((lane & 16) ? 8 : 0);
#pragma unroll
      for (int r = 0; r < 8; ++r) {
        float v = acc1[r] + bv;
        hid[(rbase + r) * HID_STRIDE + col] = f2bf(v > 0.0f ? v : 0.0f);
      }
    }
    // ---- GEMM2: acc2[32x768] += hid[32x64] @ W2e[kh:kh+64, :], two K=32 steps
    for (int ks = 0; ks < 64; ks += 32) {
      // stage W2 slice transposed: w2t[c][r] = W2e[kh+ks+r][c], c in [0,768)
      for (int i = tid; i < 32 * D_; i += 256) {
        const int r = i / D_, c = i % D_;
        w2t[c * W2T_STRIDE + r] = f2bf(W2e[(size_t)(kh + ks + r) * D_ + c]);
      }
      if (kh + ks + 32 < H_)
        __builtin_prefetch(&W2e[(size_t)(kh + ks + 32) * D_], 0, 1);
      __syncthreads();
      union { v16bf v; v4u u[2]; } A2;
      {
        const int row = mt * 16 + (lane & 15);
        const unsigned short* p =
            &hid[row * HID_STRIDE + ks + ((lane < 16) ? 0 : 8)];
        A2.u[0] = *(const v4u*)(p);
        A2.u[1] = *(const v4u*)(p + 16);
      }
#pragma unroll
      for (int j = 0; j < 12; ++j) {
        const int n = (cg + j * 4) * 16 + (lane & 15);
        union { v16bf v; v4u u[2]; } B2;
        const unsigned short* p =
            &w2t[n * W2T_STRIDE + ((lane < 16) ? 0 : 8)];
        B2.u[0] = *(const v4u*)(p);
        B2.u[1] = *(const v4u*)(p + 16);
        acc2[j] = __builtin_amdgcn_wmma_f32_16x16x32_bf16(
            false, A2.v, false, B2.v, (short)0, acc2[j], false, false);
      }
      __syncthreads();   // w2t restaged next sub-step / hid rewritten next chunk
    }
  }

  // ---- epilogue: + b2, * gate prob, atomic scatter into out ----
  const int rbase = mt * 16 + ((lane & 16) ? 8 : 0);
#pragma unroll
  for (int j = 0; j < 12; ++j) {
    const int col = (cg + j * 4) * 16 + (lane & 15);
    const float b2v = b2[(size_t)e * D_ + col];
#pragma unroll
    for (int r = 0; r < 8; ++r) {
      const int row = rbase + r;
      if (m0 + row < cnt) {
        atomicAdd(&out[(size_t)toks_s[row] * D_ + col],
                  probs_s[row] * (acc2[j][r] + b2v));
      }
    }
  }
}

// ---------------------------------------------------------------------------
extern "C" void kernel_launch(void* const* d_in, const int* in_sizes, int n_in,
                              void* d_out, int out_size, void* d_ws, size_t ws_size,
                              hipStream_t stream) {
  const float* x     = (const float*)d_in[0];
  const float* noise = (const float*)d_in[1];
  const float* Wg    = (const float*)d_in[2];
  const float* bg    = (const float*)d_in[3];
  const float* Wn    = (const float*)d_in[4];
  const float* bn    = (const float*)d_in[5];
  const float* W1    = (const float*)d_in[6];
  const float* b1    = (const float*)d_in[7];
  const float* W2    = (const float*)d_in[8];
  const float* b2    = (const float*)d_in[9];
  float* out = (float*)d_out;

  // workspace layout: counts[8] | pad to 256B | btok[E*N] | bprob[E*N]
  int*   counts = (int*)d_ws;
  int*   btok   = (int*)((char*)d_ws + 256);
  float* bprob  = (float*)((char*)d_ws + 256 + (size_t)E_ * NTOK * sizeof(int));

  // k0: zero output + counts (fresh every launch; harness only poisons once)
  zero_kernel<<<(out_size + 255) / 256, 256, 0, stream>>>(out, out_size, counts);
  // k1: gating / routing (8 tokens per 256-thread block)
  gating_kernel<<<NTOK / 8, 256, 0, stream>>>(x, noise, Wg, bg, Wn, bn,
                                              counts, btok, bprob);
  // k2: expert FFN — 128 tile-slots per expert (worst case all tokens on one
  // expert); blocks past counts[e] exit immediately.
  dim3 grid(NTOK / 32, E_);
  moe_ffn_kernel<<<grid, 256, SMEM_BYTES, stream>>>(x, W1, b1, W2, b2,
                                                    counts, btok, bprob, out);
}
